// FlexiblePDHGFront_15599321219637
// MI455X (gfx1250) — matrix operand order repaired and checked by reading, not compiled, not run on hardware
//
#include <hip/hip_runtime.h>

typedef __attribute__((ext_vector_type(16))) _Float16 v16h;
typedef __attribute__((ext_vector_type(8)))  _Float16 v8h;
typedef __attribute__((ext_vector_type(8)))  float    v8f;

#define PD_TAU   0.5f
#define PD_SIGMA 0.5f
#define PD_RES   0.5f
#define PD_J     10

// butterfly add over each 16-lane half (xor masks 1,2,4,8 never cross halves)
template<int OFF>
__device__ __forceinline__ float swadd(float v) {
  int s = __builtin_amdgcn_ds_swizzle(__float_as_int(v), OFF);
  return v + __int_as_float(s);
}
__device__ __forceinline__ float red16(float v) {
  v = swadd<0x041F>(v);   // xor 1, and 0x1f
  v = swadd<0x081F>(v);   // xor 2
  v = swadd<0x101F>(v);   // xor 4
  v = swadd<0x201F>(v);   // xor 8
  return v;               // sum over 16-lane half, broadcast to all its lanes
}

__device__ __forceinline__ v8f wmma_f16(v16h a, v16h b, v8f c) {
  return __builtin_amdgcn_wmma_f32_16x16x32_f16(
      /*neg_a=*/false, a, /*neg_b=*/false, b,
      /*c_mod=*/(short)0, c, /*reuse_a=*/false, /*reuse_b=*/false);
}

// A-fragment (16x32 f16) K index for slot i of this lane-half kh
__device__ __forceinline__ int kA(int i, int kh) {
  return ((i & 8) << 1) + kh * 8 + (i & 7);
}

// row-broadcast (v[r] = point kh*8+r, uniform across each 16-lane half)
//   -> lane-per-point (lane L gets value of point L&15).
// 7 cndmask (lane-bit masks are loop-invariant) + 1 ds_bpermute across halves.
__device__ __forceinline__ float row8_to_lane(const float (&v)[8], int lane) {
  const float a0 = (lane & 1) ? v[1] : v[0];
  const float a1 = (lane & 1) ? v[3] : v[2];
  const float a2 = (lane & 1) ? v[5] : v[4];
  const float a3 = (lane & 1) ? v[7] : v[6];
  const float b0 = (lane & 2) ? a1 : a0;
  const float b1 = (lane & 2) ? a3 : a2;
  const float c  = (lane & 4) ? b1 : b0;           // = v[lane&7] (own half)
  const int src  = (((lane & 8) << 1) + (lane & 7)) << 2;  // lane of half (L>>3)&1
  return __int_as_float(__builtin_amdgcn_ds_bpermute(src, __float_as_int(c)));
}

__global__ __launch_bounds__(256)
void FlexiblePDHGFront_kernel(
    const float* __restrict__ x,   const float* __restrict__ G,
    const float* __restrict__ h,   const float* __restrict__ W1,
    const float* __restrict__ b1,  const float* __restrict__ W2,
    const float* __restrict__ b2,  const float* __restrict__ Wmu,
    const float* __restrict__ bmu, const float* __restrict__ Wy,
    const float* __restrict__ by,  const float* __restrict__ Wp1,
    const float* __restrict__ bp1, const float* __restrict__ Wp2,
    const float* __restrict__ bp2, float* __restrict__ out, int N)
{
  // only LDS use: one-time feats2 C->A relayout, strictly intra-wave
  __shared__ __align__(16) _Float16 ldsF[8][16][32];

  const int tid  = threadIdx.x;
  const int lane = tid & 31;
  const int wave = tid >> 5;
  const int col  = lane & 15;   // E / feature column owned by this lane
  const int kh   = lane >> 4;   // lane half (row-block / K-block select)

  int tileBase = ((int)blockIdx.x * 8 + wave) * 16;
  if (tileBase + 16 > N) tileBase = (N >= 16) ? (N - 16) : 0;

  // ---- per-lane point coords (row = col for A-fragment builds) ----
  const float2 xr = ((const float2*)x)[tileBase + col];

  // x redistributed lane-per-point -> row-broadcast via bpermute (one-time)
  float px0[8], px1[8];
  #pragma unroll
  for (int r = 0; r < 8; ++r) {
    const int src = (kh * 8 + r) << 2;   // lanes 0..15 hold points 0..15
    px0[r] = __int_as_float(__builtin_amdgcn_ds_bpermute(src, __float_as_int(xr.x)));
    px1[r] = __int_as_float(__builtin_amdgcn_ds_bpermute(src, __float_as_int(xr.y)));
  }

  // geometry per lane + scalar invariants of G,h (broadcast via one-time red16)
  const float Ge0 = G[col * 2 + 0], Ge1 = G[col * 2 + 1], he = h[col];
  const float S00 = red16(Ge0 * Ge0);   // sum_e G[e,0]^2
  const float S01 = red16(Ge0 * Ge1);   // sum_e G[e,0]*G[e,1]
  const float S11 = red16(Ge1 * Ge1);   // sum_e G[e,1]^2
  const float hG0 = red16(he * Ge0);    // sum_e h[e]*G[e,0]
  const float hG1 = red16(he * Ge1);

  // ---- encoder: feats1 = relu(x@W1+b1) directly in A-fragment layout ----
  v16h a1;
  #pragma unroll
  for (int i = 0; i < 16; ++i) {
    const int k = kA(i, kh);
    const float f = fmaf(xr.x, W1[k], fmaf(xr.y, W1[32 + k], b1[k]));
    a1[i] = (_Float16)fmaxf(f, 0.0f);
  }
  // W2 as two 32x16 B fragments (cols 0-15, 16-31)
  v16h bW2a, bW2b;
  #pragma unroll
  for (int i = 0; i < 16; ++i) {
    const int k = kA(i, kh);
    bW2a[i] = (_Float16)W2[k * 32 + col];
    bW2b[i] = (_Float16)W2[k * 32 + 16 + col];
  }
  v8f cA, cB;
  { const float ba = b2[col], bb = b2[16 + col];
    #pragma unroll
    for (int r = 0; r < 8; ++r) { cA[r] = ba; cB[r] = bb; } }
  cA = wmma_f16(a1, bW2a, cA);
  cB = wmma_f16(a1, bW2b, cB);

  float f2a[8], f2b[8];
  #pragma unroll
  for (int r = 0; r < 8; ++r) {
    f2a[r] = fmaxf(cA[r], 0.0f);
    f2b[r] = fmaxf(cB[r], 0.0f);
    ldsF[wave][kh * 8 + r][col]      = (_Float16)f2a[r];
    ldsF[wave][kh * 8 + r][16 + col] = (_Float16)f2b[r];
  }
  // intra-wave cross-lane LDS dependency: stop compiler reordering the
  // following loads above the stores (hardware keeps same-wave DS in order)
  __asm__ volatile("" ::: "memory");
  // feats2 as A fragment: per-lane k-runs are contiguous -> two 16B LDS loads
  v16h a2;
  { const v8h lo = *(const v8h*)&ldsF[wave][col][kh * 8];
    const v8h hi = *(const v8h*)&ldsF[wave][col][16 + kh * 8];
    #pragma unroll
    for (int i = 0; i < 8; ++i) { a2[i] = lo[i]; a2[8 + i] = hi[i]; } }

  // ---- mu0 = relu(feats2 @ Wmu + bmu), lands in S (C-frag) layout ----
  v16h bWmu;
  #pragma unroll
  for (int i = 0; i < 16; ++i) bWmu[i] = (_Float16)Wmu[kA(i, kh) * 16 + col];
  v8f mu;
  { const float bm = bmu[col];
    v8f cm;
    #pragma unroll
    for (int r = 0; r < 8; ++r) cm[r] = bm;
    mu = wmma_f16(a2, bWmu, cm);
    #pragma unroll
    for (int r = 0; r < 8; ++r) mu[r] = fmaxf(mu[r], 0.0f); }

  // ---- y0 = tanh(feats2 @ Wy + by) via lane reductions ----
  const float wy0a = Wy[col * 2 + 0],        wy1a = Wy[col * 2 + 1];
  const float wy0b = Wy[(16 + col) * 2 + 0], wy1b = Wy[(16 + col) * 2 + 1];
  const float by0 = by[0], by1 = by[1];
  float y0[8], y1[8];
  #pragma unroll
  for (int r = 0; r < 8; ++r) {
    y0[r] = tanhf(red16(fmaf(f2a[r], wy0a, f2b[r] * wy0b)) + by0);
    y1[r] = tanhf(red16(fmaf(f2a[r], wy1a, f2b[r] * wy1b)) + by1);
  }

  // ---- a_row (S layout) + per-point scalars aG = M x - G^T h ----
  float arow[8], aG0[8], aG1[8];
  #pragma unroll
  for (int r = 0; r < 8; ++r) {
    arow[r] = fmaf(px0[r], Ge0, fmaf(px1[r], Ge1, -he));
    aG0[r]  = fmaf(px0[r], S00, fmaf(px1[r], S01, -hG0));
    aG1[r]  = fmaf(px0[r], S01, fmaf(px1[r], S11, -hG1));
  }

  // ---- carried t = mu@G (only init needs a reduction) ----
  float t0[8], t1[8];
  #pragma unroll
  for (int r = 0; r < 8; ++r) {
    t0[r] = red16(mu[r] * Ge0);
    t1[r] = red16(mu[r] * Ge1);
  }

  // ---- prox constants: fold RES into B fragment + bias ----
  v16h bWp2;
  #pragma unroll
  for (int i = 0; i < 16; ++i)
    bWp2[i] = (_Float16)(PD_RES * Wp2[kA(i, kh) * 16 + col]);
  float wp10[16], wp11[16], bp1v[16];
  #pragma unroll
  for (int i = 0; i < 16; ++i) {
    const int k = kA(i, kh);
    wp10[i] = Wp1[k]; wp11[i] = Wp1[32 + k]; bp1v[i] = bp1[k];
  }
  const float bp2e = PD_RES * bp2[col];

  // ---- 10 PDHG steps: 1 WMMA + 1 reduction group per step, no barriers ----
  for (int it = 0; it < PD_J; ++it) {
    float z0[8], z1[8];
    #pragma unroll
    for (int r = 0; r < 8; ++r) {
      // dual ascent + unit-ball projection on y (t = mu@G carried)
      const float u0 = fmaf(PD_SIGMA, t0[r], y0[r]);
      const float u1 = fmaf(PD_SIGMA, t1[r], y1[r]);
      const float inv = 1.0f / fmaxf(sqrtf(fmaf(u0, u0, u1 * u1)), 1.0f);
      const float yn0 = u0 * inv, yn1 = u1 * inv;
      y0[r] = yn0; y1[r] = yn1;
      // primal update on mu (vector part, S layout)
      mu[r] = fmaf(PD_TAU, arow[r] - fmaf(yn0, Ge0, yn1 * Ge1), mu[r]);
      // z = mu_upd@G in closed form: z = t + TAU*(aG - M y)
      z0[r] = fmaf(PD_TAU, aG0[r] - fmaf(yn0, S00, yn1 * S01), t0[r]);
      z1[r] = fmaf(PD_TAU, aG1[r] - fmaf(yn0, S01, yn1 * S11), t1[r]);
    }
    // redistribute z in-wave: cndmask tree + ds_bpermute (no LDS, no barriers)
    const float zx = row8_to_lane(z0, lane);
    const float zy = row8_to_lane(z1, lane);
    // hid = relu(z@Wp1 + bp1) built directly as A fragment
    v16h ah;
    #pragma unroll
    for (int i = 0; i < 16; ++i)
      ah[i] = (_Float16)fmaxf(fmaf(zx, wp10[i], fmaf(zy, wp11[i], bp1v[i])), 0.0f);
    // mu' = (mu + RES*bp2) + hid @ (RES*Wp2)  (single WMMA, C carries mu)
    v8f cin;
    #pragma unroll
    for (int r = 0; r < 8; ++r) cin[r] = mu[r] + bp2e;
    v8f mun = wmma_f16(ah, bWp2, cin);
    // project_mu: clamp, one reduction group for n = mun@G, carry t = n/s
    #pragma unroll
    for (int r = 0; r < 8; ++r) mun[r] = fmaxf(mun[r], 0.0f);
    #pragma unroll
    for (int r = 0; r < 8; ++r) {
      const float n0 = red16(mun[r] * Ge0);
      const float n1 = red16(mun[r] * Ge1);
      const float inv = 1.0f / fmaxf(sqrtf(fmaf(n0, n0, n1 * n1)), 1.0f);
      mu[r] = mun[r] * inv;
      t0[r] = n0 * inv;
      t1[r] = n1 * inv;
    }
  }

  // ---- final project_mu via carried t (no reductions) + store ----
  #pragma unroll
  for (int r = 0; r < 8; ++r) {
    const float m  = fmaxf(mu[r], 0.0f);
    const float vn = sqrtf(fmaf(t0[r], t0[r], t1[r] * t1[r]));
    const float mf = m / fmaxf(vn, 1.0f);
    out[(tileBase + kh * 8 + r) * 16 + col] = mf;
  }
}

extern "C" void kernel_launch(void* const* d_in, const int* in_sizes, int n_in,
                              void* d_out, int out_size, void* d_ws, size_t ws_size,
                              hipStream_t stream) {
  const float* x   = (const float*)d_in[0];
  const float* G   = (const float*)d_in[1];
  const float* h   = (const float*)d_in[2];
  const float* W1  = (const float*)d_in[3];
  const float* b1  = (const float*)d_in[4];
  const float* W2  = (const float*)d_in[5];
  const float* b2  = (const float*)d_in[6];
  const float* Wmu = (const float*)d_in[7];
  const float* bmu = (const float*)d_in[8];
  const float* Wy  = (const float*)d_in[9];
  const float* by  = (const float*)d_in[10];
  const float* Wp1 = (const float*)d_in[11];
  const float* bp1 = (const float*)d_in[12];
  const float* Wp2 = (const float*)d_in[13];
  const float* bp2 = (const float*)d_in[14];
  float* out = (float*)d_out;

  const int N = in_sizes[0] / 2;          // x is [N,2]
  const int blocks = (N + 127) / 128;     // 8 waves/block * 16 points/wave
  hipLaunchKernelGGL(FlexiblePDHGFront_kernel, dim3(blocks), dim3(256), 0, stream,
                     x, G, h, W1, b1, W2, b2, Wmu, bmu, Wy, by, Wp1, bp1, Wp2, bp2,
                     out, N);
}
